// Controller_11166914970593
// MI455X (gfx1250) — matrix operand order repaired and checked by reading, not compiled, not run on hardware
//
#include <hip/hip_runtime.h>
#include <math.h>

#define D 64
#define G 256            // 4*D gate outputs
#define KTOT 128         // [x ; h] fused K

typedef __attribute__((ext_vector_type(16))) _Float16 v16h;
typedef __attribute__((ext_vector_type(8)))  float    v8f;

__device__ __forceinline__ float sigm(float x) { return 1.0f / (1.0f + __expf(-x)); }

__device__ __forceinline__ unsigned long long splitmix(unsigned long long z) {
    z += 0x9E3779B97F4A7C15ull;
    z = (z ^ (z >> 30)) * 0xBF58476D1CE4E5B9ull;
    z = (z ^ (z >> 27)) * 0x94D049BB133111EBull;
    return z ^ (z >> 31);
}
__device__ __forceinline__ float rng_u01(int step) {
    unsigned long long r = splitmix(0x2A6A09E667F3BCC9ull + (unsigned long long)step * 0x100000001B3ull);
    return ((float)(r >> 40) + 0.5f) * (1.0f / 16777216.0f);
}

__global__ __launch_bounds__(256, 1)
void enas_controller_kernel(const float* __restrict__ embed_w,   // [6,64]
                            const float* __restrict__ w_ih,      // [256,64]
                            const float* __restrict__ w_hh,      // [256,64]
                            const float* __restrict__ b_ih,      // [256]
                            const float* __restrict__ b_hh,      // [256]
                            const float* __restrict__ w_soft,    // [5,64]
                            const float* __restrict__ b_soft,    // [5]
                            const float* __restrict__ b_soft_nl, // [5]
                            const float* __restrict__ w_attn_1,  // [64,64]
                            const float* __restrict__ w_attn_2,  // [64,64]
                            const float* __restrict__ v_attn,    // [64]
                            float* __restrict__ out)             // [42]
{
    const int t    = threadIdx.x;
    const int lane = t & 31;
    const int wave = t >> 5;

    // ---- LDS-resident model + state (fits in one WGP's LDS) ----
    __shared__ _Float16 sW[G * KTOT];     // 64 KB: row r holds [w_ih[r,0:64] | w_hh[r,0:64]] in f16
    __shared__ _Float16 sA1[D * D];       // 8 KB  w_attn_1 (f16)
    __shared__ _Float16 sA2[D * D];       // 8 KB  w_attn_2 (f16)
    __shared__ _Float16 xcatH[KTOT];      // [x ; h] stored directly in f16 (WMMA A operand)
    __shared__ float sEmbed[6 * D];
    __shared__ float sWsoft[5 * D];
    __shared__ float sBias[G];            // b_ih + b_hh
    __shared__ float sVat[D];
    __shared__ float sBs[5], sBsnl[5];
    __shared__ float hbuf[D], cbuf[D];
    __shared__ float gbuf[G];
    __shared__ float anchors[7 * D];
    __shared__ float anchw1[7 * D];
    __shared__ float hw2[D];
    __shared__ float red[D];
    __shared__ float logitsS[8];
    __shared__ int   idxS;
    __shared__ float accLP, accENT;
    __shared__ int   arcS[40];

    // ---- one-shot weight load + f16 convert ----
    for (int i = t; i < G * D; i += 256) {
        int r = i >> 6, k = i & 63;
        sW[r * KTOT + k]       = (_Float16)w_ih[i];
        sW[r * KTOT + D + k]   = (_Float16)w_hh[i];
    }
    for (int i = t; i < D * D; i += 256) { sA1[i] = (_Float16)w_attn_1[i]; sA2[i] = (_Float16)w_attn_2[i]; }
    for (int i = t; i < 6 * D; i += 256) sEmbed[i] = embed_w[i];
    for (int i = t; i < 5 * D; i += 256) sWsoft[i] = w_soft[i];
    if (t < G)  sBias[t] = b_ih[t] + b_hh[t];
    if (t < D)  { sVat[t] = v_attn[t]; hbuf[t] = 0.0f; cbuf[t] = 0.0f; xcatH[D + t] = (_Float16)0.0f; }
    if (t < 5)  { sBs[t] = b_soft[t]; sBsnl[t] = b_soft_nl[t]; }
    if (t == 0) { accLP = 0.0f; accENT = 0.0f; }
    __syncthreads();

    // ---- LSTM cell: gate GEMV [1,128] x [128,256] on WMMA, 8 waves x 2 tiles ----
    auto lstm_cell = [&]() {
        __syncthreads();
        const int khalf = (lane & 16) ? 1 : 0;

        // Build the 4 A-operand K-chunks once (f16 LDS -> registers, no converts).
        // Broadcast x into all 16 M rows: every output row of D equals x @ B.
        v16h av[4];
        #pragma unroll
        for (int kc = 0; kc < 4; ++kc) {
            const int ka = kc * 32 + khalf * 16;
            #pragma unroll
            for (int e = 0; e < 8; ++e) {
                av[kc][e]     = xcatH[ka + e];       // 16B contiguous -> ds_load_b128
                av[kc][8 + e] = xcatH[ka + 8 + e];   // 16B contiguous -> ds_load_b128
            }
        }

        #pragma unroll
        for (int tt = 0; tt < 2; ++tt) {
            const int tile = wave * 2 + tt;
            const int col  = tile * 16 + (lane & 15);
            v8f acc;
            const float bias = sBias[col];
            #pragma unroll
            for (int j = 0; j < 8; ++j) acc[j] = bias;
            #pragma unroll
            for (int kc = 0; kc < 4; ++kc) {
                const int kb = kc * 32 + khalf * 16; // B: col n = lane&15, 16 contiguous K halves
                v16h b;
                #pragma unroll
                for (int e = 0; e < 16; ++e) b[e] = sW[col * KTOT + kb + e];
                acc = __builtin_amdgcn_wmma_f32_16x16x32_f16(
                          false, av[kc], false, b, (short)0, acc, false, false);
            }
            if (lane < 16) gbuf[col] = acc[0];       // row M=0 of D tile
        }
        __syncthreads();
        if (t < D) {
            float gi = gbuf[t], gf = gbuf[D + t], gg = gbuf[2 * D + t], go = gbuf[3 * D + t];
            float cc = sigm(gf) * cbuf[t] + sigm(gi) * tanhf(gg);
            cbuf[t] = cc;
            float hh = sigm(go) * tanhf(cc);
            hbuf[t] = hh;
            xcatH[D + t] = (_Float16)hh;             // next cell's h input (single convert)
        }
        __syncthreads();
    };

    auto reduce64 = [&](float val) -> float {        // sum over t<64; all threads call
        if (t < D) red[t] = val;
        __syncthreads();
        if (t < 32) red[t] += red[t + 32];
        __syncthreads();
        if (t < 16) red[t] += red[t + 16];
        __syncthreads();
        if (t < 8)  red[t] += red[t + 8];
        __syncthreads();
        if (t < 4)  red[t] += red[t + 4];
        __syncthreads();
        if (t < 2)  red[t] += red[t + 2];
        __syncthreads();
        if (t == 0) red[0] += red[1];
        __syncthreads();
        return red[0];
    };

    int step = 0;
    for (int samp = 0; samp < 2; ++samp) {
        const bool use_bias = (samp == 0);
        int arcpos = samp * 20;
        int nanch = 0;

        if (t < D) xcatH[t] = (_Float16)sEmbed[t];   // embed(0)
        __syncthreads();

        // two warm-up cells; anchors are zeros, anchw1 = h @ W1^T
        for (int s = 0; s < 2; ++s) {
            lstm_cell();
            if (t < D) {
                anchors[nanch * D + t] = 0.0f;
                float acc1 = 0.0f;
                for (int k = 0; k < D; ++k) acc1 += hbuf[k] * (float)sA1[t * D + k];
                anchw1[nanch * D + t] = acc1;
            }
            __syncthreads();
            ++nanch;
        }

        for (int layer = 2; layer < 7; ++layer) {
            // -- sample two input-node indices via attention over anchors --
            for (int ii = 0; ii < 2; ++ii) {
                lstm_cell();
                if (t < D) {
                    float acc2 = 0.0f;
                    for (int k = 0; k < D; ++k) acc2 += hbuf[k] * (float)sA2[t * D + k];
                    hw2[t] = acc2;
                }
                __syncthreads();
                for (int a = 0; a < nanch; ++a) {
                    float e = 0.0f;
                    if (t < D) e = tanhf(anchw1[a * D + t] + hw2[t]) * sVat[t];
                    float s = reduce64(e);
                    if (t == 0) logitsS[a] = s / 5.0f + 1.1f * tanhf(s);
                    __syncthreads();
                }
                if (t == 0) {
                    float mx = -1e30f;
                    for (int a = 0; a < nanch; ++a) mx = fmaxf(mx, logitsS[a]);
                    float den = 0.0f;
                    for (int a = 0; a < nanch; ++a) den += __expf(logitsS[a] - mx);
                    float lden = __logf(den);
                    float u = rng_u01(step);
                    float cum = 0.0f; int pick = nanch - 1;
                    for (int a = 0; a < nanch; ++a) {
                        cum += __expf(logitsS[a] - mx) / den;
                        if (u < cum) { pick = a; break; }
                    }
                    float ent = 0.0f;
                    for (int a = 0; a < nanch; ++a) {
                        float ls = (logitsS[a] - mx) - lden;
                        ent -= __expf(ls) * ls;
                    }
                    accLP  += -((logitsS[pick] - mx) - lden);
                    accENT += ent;
                    idxS = pick;
                    arcS[arcpos + ii * 2] = pick;
                }
                ++step;
                __syncthreads();
                const int pick = idxS;
                if (t < D) xcatH[t] = (_Float16)anchors[pick * D + t]; // gather anchor as next embed
                __syncthreads();
            }
            // -- sample two op ids --
            for (int ii = 0; ii < 2; ++ii) {
                lstm_cell();
                for (int j = 0; j < 5; ++j) {
                    float e = (t < D) ? hbuf[t] * sWsoft[j * D + t] : 0.0f;
                    float s = reduce64(e);
                    if (t == 0) {
                        float l = s + sBs[j];
                        l = (1.1f / 2.5f) * tanhf(l / 5.0f);
                        if (use_bias) l += sBsnl[j];
                        logitsS[j] = l;
                    }
                    __syncthreads();
                }
                if (t == 0) {
                    float mx = -1e30f;
                    for (int a = 0; a < 5; ++a) mx = fmaxf(mx, logitsS[a]);
                    float den = 0.0f;
                    for (int a = 0; a < 5; ++a) den += __expf(logitsS[a] - mx);
                    float lden = __logf(den);
                    float u = rng_u01(step);
                    float cum = 0.0f; int op = 4;
                    for (int a = 0; a < 5; ++a) {
                        cum += __expf(logitsS[a] - mx) / den;
                        if (u < cum) { op = a; break; }
                    }
                    float ent = 0.0f;
                    for (int a = 0; a < 5; ++a) {
                        float ls = (logitsS[a] - mx) - lden;
                        ent -= __expf(ls) * ls;
                    }
                    accLP  += -((logitsS[op] - mx) - lden);
                    accENT += ent;
                    idxS = op;
                    arcS[arcpos + ii * 2 + 1] = op;           // [idx0, op0, idx1, op1]
                }
                ++step;
                __syncthreads();
                const int op = idxS;
                if (t < D) xcatH[t] = (_Float16)sEmbed[(op + 1) * D + t];
                __syncthreads();
            }
            // -- anchor cell --
            lstm_cell();
            if (t < D) {
                anchors[nanch * D + t] = hbuf[t];
                float acc1 = 0.0f;
                for (int k = 0; k < D; ++k) acc1 += hbuf[k] * (float)sA1[t * D + k];
                anchw1[nanch * D + t] = acc1;
                xcatH[t] = (_Float16)sEmbed[t];               // embed(0) for next step
            }
            __syncthreads();
            ++nanch;
            arcpos += 4;
        }
        // h, c persist in hbuf/cbuf for the second sampler call
    }

    __syncthreads();
    if (t == 0) {
        for (int i = 0; i < 40; ++i) out[i] = (float)arcS[i];
        out[40] = accLP;
        out[41] = accENT;
    }
}

extern "C" void kernel_launch(void* const* d_in, const int* in_sizes, int n_in,
                              void* d_out, int out_size, void* d_ws, size_t ws_size,
                              hipStream_t stream) {
    (void)in_sizes; (void)n_in; (void)out_size; (void)d_ws; (void)ws_size;
    const float* embed_w    = (const float*)d_in[0];
    const float* w_ih       = (const float*)d_in[1];
    const float* w_hh       = (const float*)d_in[2];
    const float* b_ih       = (const float*)d_in[3];
    const float* b_hh       = (const float*)d_in[4];
    const float* w_soft     = (const float*)d_in[5];
    const float* b_soft     = (const float*)d_in[6];
    const float* b_soft_nl  = (const float*)d_in[7];
    const float* w_attn_1   = (const float*)d_in[8];
    const float* w_attn_2   = (const float*)d_in[9];
    const float* v_attn     = (const float*)d_in[10];

    enas_controller_kernel<<<1, 256, 0, stream>>>(
        embed_w, w_ih, w_hh, b_ih, b_hh, w_soft, b_soft, b_soft_nl,
        w_attn_1, w_attn_2, v_attn, (float*)d_out);
}
